// AlibiEncoderBlock_9878424781197
// MI455X (gfx1250) — matrix operand order repaired and checked by reading, not compiled, not run on hardware
//
#include <hip/hip_runtime.h>
#include <hip/hip_bf16.h>
#include <math.h>

// ---------------------------------------------------------------------------
// MI455X (gfx1250) ALiBi encoder block. wave32, WMMA f16->f32.
// ---------------------------------------------------------------------------

typedef __attribute__((ext_vector_type(16))) _Float16 v16h;
typedef __attribute__((ext_vector_type(8)))  _Float16 v8h;
typedef __attribute__((ext_vector_type(8)))  float    v8f;

#define HIDS   1024
#define SEQ    2048
#define NBATCH 2
#define NHEADS 16
#define HDIM   64
#define DFFS   4096

static __device__ __forceinline__ v16h cat8(v8h lo, v8h hi) {
  return __builtin_shufflevector(lo, hi, 0,1,2,3,4,5,6,7,8,9,10,11,12,13,14,15);
}

static __device__ __forceinline__ v8f wmma_f16(v16h a, v16h b, v8f c) {
  return __builtin_amdgcn_wmma_f32_16x16x32_f16(false, a, false, b, (short)0, c,
                                                false, false);
}

// A-fragment (16x32 f16): lane = row, elems = K {base..base+15}
static __device__ __forceinline__ v16h load_afrag(const _Float16* arow, int k) {
  return *(const v16h*)(arow + k);
}
// B-fragment (32x16 f16) from transposed weights: 16 sequential K, 32B aligned
static __device__ __forceinline__ v16h load_bfrag(const _Float16* brow, int k) {
  return *(const v16h*)(brow + k);
}

// ---------------------------------------------------------------------------
// Weight transpose + f32 -> f16 convert:  W[K][N] (row major) -> Wt[N][K]
// ---------------------------------------------------------------------------
__global__ __launch_bounds__(256) void wconv_kernel(const float* __restrict__ in,
                                                    _Float16* __restrict__ out,
                                                    int K, int N) {
  __shared__ float tile[32][33];
  int k0 = blockIdx.y * 32;
  int n0 = blockIdx.x * 32;
  int tx = threadIdx.x;   // 0..31
  int ty = threadIdx.y;   // 0..7
  for (int i = ty; i < 32; i += 8)
    tile[i][tx] = in[(size_t)(k0 + i) * N + (n0 + tx)];
  __syncthreads();
  for (int i = ty; i < 32; i += 8)
    out[(size_t)(n0 + i) * K + (k0 + tx)] = (_Float16)tile[tx][i];
}

// ---------------------------------------------------------------------------
// RMSNorm: f32 [rows][HIDS] -> f16 [rows][HIDS]
// ---------------------------------------------------------------------------
__global__ __launch_bounds__(256) void rmsnorm_kernel(const float* __restrict__ x,
                                                      const float* __restrict__ w,
                                                      _Float16* __restrict__ out) {
  int row = blockIdx.x;
  const float* xr = x + (size_t)row * HIDS;
  float ss = 0.0f;
  for (int c = threadIdx.x; c < HIDS; c += blockDim.x) {
    float v = xr[c];
    ss += v * v;
  }
  #pragma unroll
  for (int m = 16; m >= 1; m >>= 1) ss += __shfl_xor(ss, m, 32);
  __shared__ float wsum[8];
  __shared__ float rtotal;
  int wid = threadIdx.x >> 5;
  if ((threadIdx.x & 31) == 0) wsum[wid] = ss;
  __syncthreads();
  if (threadIdx.x == 0) {
    float t = 0.0f;
    #pragma unroll
    for (int i = 0; i < 8; i++) t += wsum[i];
    rtotal = rsqrtf(t * (1.0f / HIDS) + 1e-6f);
  }
  __syncthreads();
  float rinv = rtotal;
  for (int c = threadIdx.x; c < HIDS; c += blockDim.x)
    out[(size_t)row * HIDS + c] = (_Float16)(xr[c] * rinv * w[c]);
}

// ---------------------------------------------------------------------------
// WMMA GEMM: C[M][N] = epilogue(A[M][K] @ Bt[N][K]^T)
//   8 waves/block (4x2), each wave computes a 32x32 tile (2x2 WMMA tiles),
//   block tile 128x64. K stepped by 64 with branch-free ping-pong fragment
//   sets (peeled tail), so 8 b128 loads are always in flight behind 4 WMMAs.
//   Requires K % 64 == 0 (holds: K = 1024 or 4096).
//   epilogue: v = acc + bias[n]; if gelu: exact GELU; v *= alpha; v += res[m][n]
// ---------------------------------------------------------------------------
__global__ __launch_bounds__(256) void gemm_wmma_kernel(
    const _Float16* __restrict__ A,   // [M][K] row major f16
    const _Float16* __restrict__ Bt,  // [N][K] row major f16 (B transposed)
    const float* __restrict__ bias,   // [N] or nullptr
    const float* __restrict__ res,    // [M][N] f32 or nullptr
    float alpha, int gelu,
    float* __restrict__ outf,         // [M][N] f32 or nullptr
    _Float16* __restrict__ outh,      // [M][N] f16 or nullptr
    int M, int N, int K) {
  int lane = threadIdx.x & 31;
  int wave = threadIdx.x >> 5;
  int hlf  = lane >> 4;
  int l16  = lane & 15;
  int rowbase = blockIdx.y * 128 + (wave >> 1) * 32;
  int colbase = blockIdx.x * 64 + (wave & 1) * 32;
  (void)M;

  const _Float16* arow0 = A  + (size_t)(rowbase + l16)      * K + (hlf ? 16 : 0);
  const _Float16* arow1 = A  + (size_t)(rowbase + 16 + l16) * K + (hlf ? 16 : 0);
  const _Float16* brow0 = Bt + (size_t)(colbase + l16)      * K + (hlf ? 16 : 0);
  const _Float16* brow1 = Bt + (size_t)(colbase + 16 + l16) * K + (hlf ? 16 : 0);

  v8f acc00 = {}, acc01 = {}, acc10 = {}, acc11 = {};

  v16h a0c = load_afrag(arow0, 0);
  v16h a1c = load_afrag(arow1, 0);
  v16h b0c = load_bfrag(brow0, 0);
  v16h b1c = load_bfrag(brow1, 0);
  v16h a0n, a1n, b0n, b1n;

  int k = 0;
  for (; k < K - 64; k += 64) {
    // load "next" set (k+32) behind the current WMMAs
    a0n = load_afrag(arow0, k + 32);
    a1n = load_afrag(arow1, k + 32);
    b0n = load_bfrag(brow0, k + 32);
    b1n = load_bfrag(brow1, k + 32);
    __builtin_prefetch(arow0 + k + 128, 0, 3);
    __builtin_prefetch(arow1 + k + 128, 0, 3);
    __builtin_prefetch(brow0 + k + 128, 0, 3);
    __builtin_prefetch(brow1 + k + 128, 0, 3);
    acc00 = wmma_f16(a0c, b0c, acc00);
    acc01 = wmma_f16(a0c, b1c, acc01);
    acc10 = wmma_f16(a1c, b0c, acc10);
    acc11 = wmma_f16(a1c, b1c, acc11);
    // reload "current" set (k+64): SSA redefinition, no register rotation
    a0c = load_afrag(arow0, k + 64);
    a1c = load_afrag(arow1, k + 64);
    b0c = load_bfrag(brow0, k + 64);
    b1c = load_bfrag(brow1, k + 64);
    acc00 = wmma_f16(a0n, b0n, acc00);
    acc01 = wmma_f16(a0n, b1n, acc01);
    acc10 = wmma_f16(a1n, b0n, acc10);
    acc11 = wmma_f16(a1n, b1n, acc11);
  }
  // peeled tail: k == K - 64
  a0n = load_afrag(arow0, k + 32);
  a1n = load_afrag(arow1, k + 32);
  b0n = load_bfrag(brow0, k + 32);
  b1n = load_bfrag(brow1, k + 32);
  acc00 = wmma_f16(a0c, b0c, acc00);
  acc01 = wmma_f16(a0c, b1c, acc01);
  acc10 = wmma_f16(a1c, b0c, acc10);
  acc11 = wmma_f16(a1c, b1c, acc11);
  acc00 = wmma_f16(a0n, b0n, acc00);
  acc01 = wmma_f16(a0n, b1n, acc01);
  acc10 = wmma_f16(a1n, b0n, acc10);
  acc11 = wmma_f16(a1n, b1n, acc11);

  float bv0 = bias ? bias[colbase + l16]      : 0.0f;
  float bv1 = bias ? bias[colbase + 16 + l16] : 0.0f;
  v8f accs[2][2] = {{acc00, acc01}, {acc10, acc11}};
  #pragma unroll
  for (int i = 0; i < 2; i++) {
    #pragma unroll
    for (int j = 0; j < 2; j++) {
      int n = colbase + j * 16 + l16;
      float bv = j ? bv1 : bv0;
      #pragma unroll
      for (int r = 0; r < 8; r++) {
        int mr = rowbase + i * 16 + r + 8 * hlf;
        float v = accs[i][j][r] + bv;
        if (gelu) v = 0.5f * v * (1.0f + erff(v * 0.70710678118654752f));
        v *= alpha;
        size_t oidx = (size_t)mr * N + n;
        if (res)  v += res[oidx];
        if (outf) outf[oidx] = v;
        if (outh) outh[oidx] = (_Float16)v;
      }
    }
  }
}

// ---------------------------------------------------------------------------
// V transpose: vbuf f16 [b][s][h*64+d] -> vt f16 [b][h][d][s]
// ---------------------------------------------------------------------------
__global__ __launch_bounds__(256) void vtrans_kernel(const _Float16* __restrict__ vbuf,
                                                     _Float16* __restrict__ vt) {
  unsigned idx = blockIdx.x * 256u + threadIdx.x;  // < 2*2048*1024
  int d = idx & 63;
  int h = (idx >> 6) & 15;
  int s = (idx >> 10) & 2047;
  int b = idx >> 21;
  vt[((size_t)(b * NHEADS + h) * HDIM + d) * SEQ + s] = vbuf[idx];
}

// ---------------------------------------------------------------------------
// Flash attention with ALiBi. One wave = one 16-query tile of one (b,h).
// q,k: f16 [b*S][HIDS] (head offset h*64). vt: f16 [b][h][64][S].
// out: f16 [b*S][HIDS].
// ---------------------------------------------------------------------------
__global__ __launch_bounds__(128) void attn_kernel(const _Float16* __restrict__ q,
                                                   const _Float16* __restrict__ kmat,
                                                   const _Float16* __restrict__ vt,
                                                   _Float16* __restrict__ out) {
  int lane = threadIdx.x & 31;
  int wave = threadIdx.x >> 5;
  int hlf  = lane >> 4;
  int l16  = lane & 15;
  int h = blockIdx.y;
  int b = blockIdx.z;
  int qbase = (blockIdx.x * 4 + wave) * 16;
  const float scale = 0.125f;                     // HD^-0.5
  float slope = exp2f(-0.5f * (float)(h + 1));    // ALiBi slope, HEADS=16

  __shared__ _Float16 ptile_all[4][16][32];
  _Float16 (*pt)[32] = ptile_all[wave];

  // --- Q fragments (A layout), d-steps 0 and 32 ---
  const _Float16* qrow =
      q + ((size_t)(b * SEQ + qbase + l16) * HIDS + h * HDIM) + (hlf ? 16 : 0);
  v16h qf0 = *(const v16h*)(qrow);
  v16h qf1 = *(const v16h*)(qrow + 32);

  const _Float16* kbase_ptr = kmat + ((size_t)(b * SEQ) * HIDS + h * HDIM);
  const _Float16* vbase_ptr = vt + (size_t)(b * NHEADS + h) * HDIM * SEQ;

  float mrun[8], lrun[8];
  v8f oacc[4] = {};
  #pragma unroll
  for (int r = 0; r < 8; r++) { mrun[r] = -1e30f; lrun[r] = 0.0f; }

  for (int kb = 0; kb < SEQ; kb += 32) {
    int key0 = kb + l16;        // column for score subtile 0
    int key1 = kb + 16 + l16;   // column for score subtile 1
    // --- scores: S = Q K^T, two WMMAs per subtile over d-steps ---
    v8f s0 = {}, s1 = {};
    {
      const _Float16* k0p = kbase_ptr + (size_t)key0 * HIDS + (hlf ? 16 : 0);
      const _Float16* k1p = kbase_ptr + (size_t)key1 * HIDS + (hlf ? 16 : 0);
      s0 = wmma_f16(qf0, *(const v16h*)(k0p),      s0);
      s0 = wmma_f16(qf1, *(const v16h*)(k0p + 32), s0);
      s1 = wmma_f16(qf0, *(const v16h*)(k1p),      s1);
      s1 = wmma_f16(qf1, *(const v16h*)(k1p + 32), s1);
    }
    // --- online softmax over this 32-key chunk ---
    float p0[8], p1[8], corr[8];
    #pragma unroll
    for (int r = 0; r < 8; r++) {
      int qi = qbase + r + 8 * hlf;
      float v0 = s0[r] * scale - slope * fabsf((float)(qi - key0));
      float v1 = s1[r] * scale - slope * fabsf((float)(qi - key1));
      float cm = fmaxf(v0, v1);
      #pragma unroll
      for (int msk = 1; msk < 16; msk <<= 1) cm = fmaxf(cm, __shfl_xor(cm, msk, 32));
      float mnew = fmaxf(mrun[r], cm);
      float c  = __expf(mrun[r] - mnew);
      float e0 = __expf(v0 - mnew);
      float e1 = __expf(v1 - mnew);
      float rs = e0 + e1;
      #pragma unroll
      for (int msk = 1; msk < 16; msk <<= 1) rs += __shfl_xor(rs, msk, 32);
      lrun[r] = lrun[r] * c + rs;
      mrun[r] = mnew;
      corr[r] = c;
      p0[r] = e0;
      p1[r] = e1;
    }
    #pragma unroll
    for (int t = 0; t < 4; t++)
      #pragma unroll
      for (int r = 0; r < 8; r++) oacc[t][r] *= corr[r];

    // --- relayout P: C layout -> LDS -> A layout fragment ---
    #pragma unroll
    for (int r = 0; r < 8; r++) {
      pt[r + 8 * hlf][l16]      = (_Float16)p0[r];
      pt[r + 8 * hlf][16 + l16] = (_Float16)p1[r];
    }
    asm volatile("s_wait_dscnt 0x0" ::: "memory");
    v8h plo = *(const v8h*)(&pt[l16][hlf ? 16 : 0]);
    v8h phi = *(const v8h*)(&pt[l16][(hlf ? 16 : 0) + 8]);
    v16h pf = cat8(plo, phi);

    // --- O += P @ V : 4 dim-tiles of 16 ---
    #pragma unroll
    for (int t = 0; t < 4; t++) {
      const _Float16* vp =
          vbase_ptr + (size_t)(t * 16 + l16) * SEQ + kb + (hlf ? 16 : 0);
      oacc[t] = wmma_f16(pf, *(const v16h*)vp, oacc[t]);
    }
  }

  #pragma unroll
  for (int r = 0; r < 8; r++) lrun[r] = 1.0f / lrun[r];
  #pragma unroll
  for (int t = 0; t < 4; t++)
    #pragma unroll
    for (int r = 0; r < 8; r++) {
      int qi = qbase + r + 8 * hlf;
      out[(size_t)(b * SEQ + qi) * HIDS + h * HDIM + t * 16 + l16] =
          (_Float16)(oacc[t][r] * lrun[r]);
    }
}

// ---------------------------------------------------------------------------
// Host orchestration
// ---------------------------------------------------------------------------
extern "C" void kernel_launch(void* const* d_in, const int* in_sizes, int n_in,
                              void* d_out, int out_size, void* d_ws, size_t ws_size,
                              hipStream_t stream) {
  (void)in_sizes; (void)n_in; (void)out_size; (void)ws_size;
  const float* x   = (const float*)d_in[0];
  const float* n1w = (const float*)d_in[1];
  const float* Wq  = (const float*)d_in[2];
  const float* bq  = (const float*)d_in[3];
  const float* Wk  = (const float*)d_in[4];
  const float* bk  = (const float*)d_in[5];
  const float* Wv  = (const float*)d_in[6];
  const float* Wo  = (const float*)d_in[7];
  const float* n2w = (const float*)d_in[8];
  const float* W1  = (const float*)d_in[9];
  const float* b1  = (const float*)d_in[10];
  const float* W2  = (const float*)d_in[11];
  const float* b2  = (const float*)d_in[12];

  char* wsp = (char*)d_ws;
  auto take = [&](size_t bytes) -> void* {
    void* p = (void*)wsp;
    wsp += (bytes + 255) & ~(size_t)255;
    return p;
  };
  const size_t TOK = (size_t)NBATCH * SEQ;  // 4096 tokens

  _Float16* Wqt  = (_Float16*)take((size_t)HIDS * HIDS * 2);
  _Float16* Wkt  = (_Float16*)take((size_t)HIDS * HIDS * 2);
  _Float16* Wvt  = (_Float16*)take((size_t)HIDS * HIDS * 2);
  _Float16* Wot  = (_Float16*)take((size_t)HIDS * HIDS * 2);
  _Float16* W1t  = (_Float16*)take((size_t)HIDS * DFFS * 2);
  _Float16* W2t  = (_Float16*)take((size_t)DFFS * HIDS * 2);
  _Float16* xn   = (_Float16*)take(TOK * HIDS * 2);
  _Float16* qb   = (_Float16*)take(TOK * HIDS * 2);
  _Float16* kb   = (_Float16*)take(TOK * HIDS * 2);
  _Float16* vb   = (_Float16*)take(TOK * HIDS * 2);
  _Float16* vtb  = (_Float16*)take(TOK * HIDS * 2);
  _Float16* ab   = (_Float16*)take(TOK * HIDS * 2);
  float*    x1   = (float*)   take(TOK * HIDS * 4);
  _Float16* xn2  = (_Float16*)take(TOK * HIDS * 2);
  _Float16* hb   = (_Float16*)take(TOK * DFFS * 2);

  dim3 tblk(32, 8);
  // Weight convert+transpose: W[K][N] -> Wt[N][K] f16
  wconv_kernel<<<dim3(HIDS / 32, HIDS / 32), tblk, 0, stream>>>(Wq, Wqt, HIDS, HIDS);
  wconv_kernel<<<dim3(HIDS / 32, HIDS / 32), tblk, 0, stream>>>(Wk, Wkt, HIDS, HIDS);
  wconv_kernel<<<dim3(HIDS / 32, HIDS / 32), tblk, 0, stream>>>(Wv, Wvt, HIDS, HIDS);
  wconv_kernel<<<dim3(HIDS / 32, HIDS / 32), tblk, 0, stream>>>(Wo, Wot, HIDS, HIDS);
  wconv_kernel<<<dim3(DFFS / 32, HIDS / 32), tblk, 0, stream>>>(W1, W1t, HIDS, DFFS);
  wconv_kernel<<<dim3(HIDS / 32, DFFS / 32), tblk, 0, stream>>>(W2, W2t, DFFS, HIDS);

  // norm1
  rmsnorm_kernel<<<(int)TOK, 256, 0, stream>>>(x, n1w, xn);

  // Q, K, V projections (block tile 128x64)
  dim3 g1(HIDS / 64, (int)TOK / 128);
  gemm_wmma_kernel<<<g1, 256, 0, stream>>>(xn, Wqt, bq, nullptr, 1.0f, 0,
                                           nullptr, qb, (int)TOK, HIDS, HIDS);
  gemm_wmma_kernel<<<g1, 256, 0, stream>>>(xn, Wkt, bk, nullptr, 1.0f, 0,
                                           nullptr, kb, (int)TOK, HIDS, HIDS);
  gemm_wmma_kernel<<<g1, 256, 0, stream>>>(xn, Wvt, nullptr, nullptr, 1.0f, 0,
                                           nullptr, vb, (int)TOK, HIDS, HIDS);

  // V transpose to [b][h][d][s]
  vtrans_kernel<<<(int)(TOK * HIDS / 256), 256, 0, stream>>>(vb, vtb);

  // attention
  attn_kernel<<<dim3(SEQ / 64, NHEADS, NBATCH), 128, 0, stream>>>(qb, kb, vtb, ab);

  // O projection + residual (f32)
  gemm_wmma_kernel<<<g1, 256, 0, stream>>>(ab, Wot, nullptr, x, 1.0f, 0,
                                           x1, nullptr, (int)TOK, HIDS, HIDS);

  // norm2
  rmsnorm_kernel<<<(int)TOK, 256, 0, stream>>>(x1, n2w, xn2);

  // FFN1 (+bias, exact GELU) -> f16
  dim3 g2(DFFS / 64, (int)TOK / 128);
  gemm_wmma_kernel<<<g2, 256, 0, stream>>>(xn2, W1t, b1, nullptr, 1.0f, 1,
                                           nullptr, hb, (int)TOK, DFFS, HIDS);

  // FFN2: out = x1 + 0.5*(h@W2 + b2) -> f32 d_out
  gemm_wmma_kernel<<<g1, 256, 0, stream>>>(hb, W2t, b2, x1, 0.5f, 0,
                                           (float*)d_out, nullptr, (int)TOK, HIDS, DFFS);
}
